// KDALayer_18227841204314
// MI455X (gfx1250) — compile-verified, matrix operand order checked by reading, not run on hardware
//
#include <hip/hip_runtime.h>
#include <hip/hip_bf16.h>
#include <math.h>

#define BDIM 2
#define TDIM 2048
#define DDIM 2048
#define HDIM 16
#define KDIM 128
#define VDIM 128
#define MROWS (BDIM*TDIM)   // 4096

typedef _Float16 f16;
typedef _Float16 v16h __attribute__((ext_vector_type(16)));
typedef _Float16 v8h  __attribute__((ext_vector_type(8)));
typedef float    v8f  __attribute__((ext_vector_type(8)));
typedef int      v4i  __attribute__((ext_vector_type(4)));

// ---------- CDNA5 async global->LDS helpers (builtin if present, else inline asm) ----------
__device__ __forceinline__ void async_copy_b128(const f16* g, f16* l) {
#if defined(__HIP_DEVICE_COMPILE__) && __has_builtin(__builtin_amdgcn_global_load_async_to_lds_b128)
  __builtin_amdgcn_global_load_async_to_lds_b128(
      (__attribute__((address_space(1))) v4i*)g,
      (__attribute__((address_space(3))) v4i*)l, 0, 0);
#else
  unsigned laddr = (unsigned)(unsigned long long)(__attribute__((address_space(3))) f16*)l;
  asm volatile("global_load_async_to_lds_b128 %0, %1, off"
               :: "v"(laddr), "v"(g) : "memory");
#endif
}

__device__ __forceinline__ void wait_async0() {
#if defined(__HIP_DEVICE_COMPILE__) && __has_builtin(__builtin_amdgcn_s_wait_asynccnt)
  __builtin_amdgcn_s_wait_asynccnt(0);
#else
  asm volatile("s_wait_asynccnt 0" ::: "memory");
#endif
}

// ---------------- f32 -> f16 convert ----------------
__global__ void cvt_f32_f16(const float* __restrict__ in, f16* __restrict__ out, int n) {
  int i = blockIdx.x * 256 + threadIdx.x;
  if (i < n) out[i] = (f16)in[i];
}

__device__ __forceinline__ v16h frag16(v8h lo, v8h hi) {
  return __builtin_shufflevector(lo, hi, 0,1,2,3,4,5,6,7,8,9,10,11,12,13,14,15);
}

// ---------------- simple WMMA GEMM (small N):  C[M,N] = A[M,Kd] * B[N,Kd]^T ----------------
__global__ void gemm_wmma_nt(const f16* __restrict__ A, const f16* __restrict__ Bm,
                             float* __restrict__ C, int M, int N, int Kd) {
  const int lane = threadIdx.x & 31;
  const int wid  = blockIdx.x * 8 + (threadIdx.x >> 5);
  const int ntiles = N >> 4;
  const int total  = (M >> 4) * ntiles;
  if (wid >= total) return;
  const int tm = wid / ntiles;
  const int tn = wid % ntiles;
  const int r  = lane & 15;
  const int hi = lane >> 4;
  const int koff = hi * 8;     // lanes 0-15: K {0..7,16..23}; lanes 16-31: K {8..15,24..31}

  const f16* arow = A  + (size_t)(tm * 16 + r) * Kd + koff;
  const f16* brow = Bm + (size_t)(tn * 16 + r) * Kd + koff;

  v8f acc = {};
  for (int k0 = 0; k0 < Kd; k0 += 32) {
    v16h av = frag16(*(const v8h*)(arow + k0), *(const v8h*)(arow + k0 + 16));
    v16h bv = frag16(*(const v8h*)(brow + k0), *(const v8h*)(brow + k0 + 16));
    acc = __builtin_amdgcn_wmma_f32_16x16x32_f16(false, av, false, bv,
                                                 (short)0, acc, false, false);
  }
  float* cbase = C + (size_t)(tm * 16 + hi * 8) * N + tn * 16 + r;
  #pragma unroll
  for (int i = 0; i < 8; ++i) cbase[(size_t)i * N] = acc[i];
}

// ---------------- tiled WMMA GEMM: 128(M) x 128(N) block tile, K-step 64, 8 waves ----------
// B panel (128 x 64 f16 = 16KB) staged to LDS via async-to-LDS, double buffered (32KB).
// Wave w owns a 16x128 strip: A fragment reused across 8 N-tiles; 16 WMMAs per barrier.
__global__ void gemm_wmma_tiled(const f16* __restrict__ A, const f16* __restrict__ Bm,
                                float* __restrict__ C, int M, int N, int Kd) {
  __shared__ __align__(16) f16 bbuf[2][128 * 64];

  const int tid  = threadIdx.x;       // 0..255
  const int lane = tid & 31;
  const int wv   = tid >> 5;          // 0..7
  const int nb   = N >> 7;            // N/128 blocks
  const int nblk = blockIdx.x % nb;
  const int mblk = blockIdx.x / nb;

  const int r    = lane & 15;
  const int hi   = lane >> 4;
  const int koff = hi * 8;

  const f16* arow = A + (size_t)(mblk * 128 + wv * 16 + r) * Kd + koff;

  // staging map: 16KB panel = 1024 x 16B chunks; 256 threads x 4 chunks
  // chunk j for thread: row = (tid>>3) + 32*j, seg = tid&7  (row stride = 64 f16 = 128B)
  const int srow = tid >> 3;          // 0..31
  const int sseg = tid & 7;           // 0..7
  const f16* gpanel = Bm + (size_t)(nblk * 128 + srow) * Kd + sseg * 8;
  const size_t grs  = (size_t)32 * Kd;              // +32 rows in global
  f16* lp0 = &bbuf[0][srow * 64 + sseg * 8];
  f16* lp1 = &bbuf[1][srow * 64 + sseg * 8];

  v8f acc[8];
  #pragma unroll
  for (int i = 0; i < 8; ++i) acc[i] = (v8f){};

  // preload first panel into buffer 0
  #pragma unroll
  for (int j = 0; j < 4; ++j)
    async_copy_b128(gpanel + (size_t)j * grs, lp0 + j * 32 * 64);
  wait_async0();
  __syncthreads();

  int cur = 0;
  for (int k0 = 0; k0 < Kd; k0 += 64) {
    // prefetch next panel into the other buffer
    if (k0 + 64 < Kd) {
      f16* lpn = cur ? lp0 : lp1;
      #pragma unroll
      for (int j = 0; j < 4; ++j)
        async_copy_b128(gpanel + (size_t)j * grs + k0 + 64, lpn + j * 32 * 64);
    }

    const f16* lb = &bbuf[cur][0];
    #pragma unroll
    for (int kk = 0; kk < 64; kk += 32) {
      v16h av = frag16(*(const v8h*)(arow + k0 + kk),
                       *(const v8h*)(arow + k0 + kk + 16));
      #pragma unroll
      for (int nt = 0; nt < 8; ++nt) {
        const f16* p = lb + (nt * 16 + r) * 64 + kk + koff;
        v16h bv = frag16(*(const v8h*)p, *(const v8h*)(p + 16));
        acc[nt] = __builtin_amdgcn_wmma_f32_16x16x32_f16(false, av, false, bv,
                                                         (short)0, acc[nt], false, false);
      }
    }

    wait_async0();      // next panel landed
    __syncthreads();    // everyone done reading current panel
    cur ^= 1;
  }

  // store 8 tiles; C/D layout: vgpr i -> M = hi*8 + i ; N = (lane&15)
  const size_t crow0 = (size_t)(mblk * 128 + wv * 16 + hi * 8);
  float* cb = C + crow0 * N + nblk * 128 + r;
  #pragma unroll
  for (int nt = 0; nt < 8; ++nt) {
    #pragma unroll
    for (int i = 0; i < 8; ++i)
      cb[(size_t)i * N + nt * 16] = acc[nt][i];
  }
}

// ---------------- causal depthwise conv(4) + SiLU (+ optional L2 norm) ----------------
__global__ void conv_silu(const float* __restrict__ in, const float* __restrict__ w,
                          float* __restrict__ out, int do_norm) {
  const int blk = blockIdx.x;          // (b*T + t)*H + h
  const int h   = blk % HDIM;
  const int bt  = blk / HDIM;
  const int t   = bt % TDIM;
  const int c   = threadIdx.x;
  const long long strideT = (long long)HDIM * 128;
  const long long base = (long long)blk * 128 + c;
  const float* wp = w + ((size_t)h * 128 + c) * 4;

  float acc = 0.f;
  #pragma unroll
  for (int i = 0; i < 4; ++i) {
    int tt = t - 3 + i;
    float xv = (tt >= 0) ? in[base + (long long)(i - 3) * strideT] : 0.f;
    acc += xv * wp[i];
  }
  acc = acc / (1.f + __expf(-acc));    // SiLU

  if (do_norm) {
    __shared__ float red[128];
    red[c] = acc * acc;
    __syncthreads();
    for (int s = 64; s > 0; s >>= 1) {
      if (c < s) red[c] += red[c + s];
      __syncthreads();
    }
    float nrm = sqrtf(red[0]);
    acc = acc / fmaxf(nrm, 1e-12f);
  }
  out[base] = acc;
}

// ---------------- decay = exp(-exp(A_log[h]) * softplus(g_raw + dt_bias)) (in place) ----------------
__global__ void decay_kernel(float* __restrict__ graw, const float* __restrict__ A_log,
                             const float* __restrict__ dt_bias, int n) {
  int i = blockIdx.x * 256 + threadIdx.x;
  if (i >= n) return;
  int hk = i % (HDIM * KDIM);
  int h  = hk / KDIM;
  float z  = graw[i] + dt_bias[hk];
  float sp = (z > 20.f) ? z : log1pf(__expf(z));
  graw[i]  = __expf(-__expf(A_log[h]) * sp);
}

// ---------------- KDA delta-rule scan ----------------
__global__ void kda_scan(const float* __restrict__ q, const float* __restrict__ k,
                         const float* __restrict__ v, const float* __restrict__ dec,
                         const float* __restrict__ beta_raw, float* __restrict__ y) {
  const int b  = blockIdx.x / HDIM;
  const int h  = blockIdx.x % HDIM;
  const int vc = threadIdx.x;
  __shared__ float kl[KDIM], ql[KDIM], dl[KDIM];

  float S[KDIM];
  #pragma unroll
  for (int i = 0; i < KDIM; ++i) S[i] = 0.f;

  for (int t = 0; t < TDIM; ++t) {
    const size_t base = ((size_t)(b * TDIM + t) * HDIM + h) * (size_t)KDIM;
    ql[vc] = q[base + vc];
    kl[vc] = k[base + vc];
    dl[vc] = dec[base + vc];
    float vt = v[base + vc];
    float br = beta_raw[(size_t)(b * TDIM + t) * HDIM + h];
    float bt = 1.f / (1.f + __expf(-br));
    __syncthreads();

    float acc = 0.f;
    #pragma unroll
    for (int kk = 0; kk < KDIM; ++kk) { S[kk] *= dl[kk]; acc = fmaf(kl[kk], S[kk], acc); }
    float werr = bt * (vt - acc);
    float o = 0.f;
    #pragma unroll
    for (int kk = 0; kk < KDIM; ++kk) { S[kk] = fmaf(werr, kl[kk], S[kk]); o = fmaf(ql[kk], S[kk], o); }

    y[base + vc] = o;
    __syncthreads();
  }
}

// ---------------- RMS-norm * o_norm_w * sigmoid(gate) -> f16 ----------------
__global__ void epilogue(const float* __restrict__ y, const float* __restrict__ gate_raw,
                         const float* __restrict__ bg, const float* __restrict__ onw,
                         f16* __restrict__ y2h) {
  const int blk = blockIdx.x;     // (b*T + t)*H + h
  const int h   = blk % HDIM;
  const int c   = threadIdx.x;
  const size_t idx = (size_t)blk * VDIM + c;
  float yv = y[idx];

  __shared__ float red[VDIM];
  red[c] = yv * yv;
  __syncthreads();
  for (int s = 64; s > 0; s >>= 1) {
    if (c < s) red[c] += red[c + s];
    __syncthreads();
  }
  float scale = rsqrtf(red[0] * (1.f / VDIM) + 1.1920929e-07f);
  float g = gate_raw[idx] + bg[h * VDIM + c];
  float r = yv * scale * onw[c] / (1.f + __expf(-g));
  y2h[idx] = (f16)r;
}

// ---------------- launch ----------------
extern "C" void kernel_launch(void* const* d_in, const int* in_sizes, int n_in,
                              void* d_out, int out_size, void* d_ws, size_t ws_size,
                              hipStream_t stream) {
  const float* x     = (const float*)d_in[0];
  const float* Wq    = (const float*)d_in[1];
  const float* Wk    = (const float*)d_in[2];
  const float* Wv    = (const float*)d_in[3];
  const float* Wf1   = (const float*)d_in[4];
  const float* Wf2   = (const float*)d_in[5];
  const float* Wb    = (const float*)d_in[6];
  const float* Wg1   = (const float*)d_in[7];
  const float* Wg2   = (const float*)d_in[8];
  const float* bg    = (const float*)d_in[9];
  const float* onw   = (const float*)d_in[10];
  const float* Wout  = (const float*)d_in[11];
  const float* A_log = (const float*)d_in[12];
  const float* dtb   = (const float*)d_in[13];
  const float* qcw   = (const float*)d_in[14];
  const float* kcw   = (const float*)d_in[15];
  const float* vcw   = (const float*)d_in[16];
  float* out = (float*)d_out;

  size_t off = 0;
  auto alloc = [&](size_t bytes) { void* p = (char*)d_ws + off; off += (bytes + 255) & ~(size_t)255; return p; };
  const size_t NE_X  = (size_t)MROWS * DDIM;        // 8M
  const size_t NE_HK = (size_t)MROWS * HDIM * KDIM; // 8M

  f16*  xh    = (f16*)alloc(NE_X * 2);
  f16*  wqh   = (f16*)alloc((size_t)HDIM*KDIM*DDIM * 2);
  f16*  wkh   = (f16*)alloc((size_t)HDIM*KDIM*DDIM * 2);
  f16*  wvh   = (f16*)alloc((size_t)HDIM*VDIM*DDIM * 2);
  f16*  wouth = (f16*)alloc((size_t)DDIM*HDIM*VDIM * 2);
  f16*  wf1h  = (f16*)alloc((size_t)VDIM*DDIM * 2);
  f16*  wf2h  = (f16*)alloc((size_t)HDIM*KDIM*VDIM * 2);
  f16*  wbh   = (f16*)alloc((size_t)HDIM*DDIM * 2);
  f16*  wg1h  = (f16*)alloc((size_t)VDIM*DDIM * 2);
  f16*  wg2h  = (f16*)alloc((size_t)HDIM*VDIM*VDIM * 2);
  float* qraw = (float*)alloc(NE_HK * 4);
  float* kraw = (float*)alloc(NE_HK * 4);
  float* vraw = (float*)alloc(NE_HK * 4);
  float* qn   = (float*)alloc(NE_HK * 4);
  float* kn   = (float*)alloc(NE_HK * 4);
  float* vn   = (float*)alloc(NE_HK * 4);
  float* graw = (float*)alloc(NE_HK * 4);   // becomes decay in place
  float* betar= (float*)alloc((size_t)MROWS*HDIM * 4);
  float* f1   = (float*)alloc((size_t)MROWS*VDIM * 4);
  f16*  f1h   = (f16*)alloc((size_t)MROWS*VDIM * 2);
  float* g1   = (float*)alloc((size_t)MROWS*VDIM * 4);
  f16*  g1h   = (f16*)alloc((size_t)MROWS*VDIM * 2);
  float* gater= (float*)alloc(NE_HK * 4);
  float* ybuf = (float*)alloc(NE_HK * 4);
  f16*  y2h   = (f16*)alloc(NE_HK * 2);
  (void)ws_size; (void)in_sizes; (void)n_in; (void)out_size;

  auto cvt = [&](const float* src, f16* dst, size_t n) {
    cvt_f32_f16<<<(unsigned)((n + 255) / 256), 256, 0, stream>>>(src, dst, (int)n);
  };
  auto gemm_big = [&](const f16* A, const f16* Bm, float* C, int M, int N, int Kd) {
    int blocks = (M / 128) * (N / 128);
    gemm_wmma_tiled<<<blocks, 256, 0, stream>>>(A, Bm, C, M, N, Kd);
  };
  auto gemm_small = [&](const f16* A, const f16* Bm, float* C, int M, int N, int Kd) {
    int tiles = (M >> 4) * (N >> 4);
    gemm_wmma_nt<<<(tiles + 7) / 8, 256, 0, stream>>>(A, Bm, C, M, N, Kd);
  };

  // 1) precision converts
  cvt(x, xh, NE_X);
  cvt(Wq, wqh, (size_t)HDIM*KDIM*DDIM);
  cvt(Wk, wkh, (size_t)HDIM*KDIM*DDIM);
  cvt(Wv, wvh, (size_t)HDIM*VDIM*DDIM);
  cvt(Wout, wouth, (size_t)DDIM*HDIM*VDIM);
  cvt(Wf1, wf1h, (size_t)VDIM*DDIM);
  cvt(Wf2, wf2h, (size_t)HDIM*KDIM*VDIM);
  cvt(Wb, wbh, (size_t)HDIM*DDIM);
  cvt(Wg1, wg1h, (size_t)VDIM*DDIM);
  cvt(Wg2, wg2h, (size_t)HDIM*VDIM*VDIM);

  // 2) projections (WMMA, async-LDS tiled)
  gemm_big(xh, wqh, qraw, MROWS, HDIM*KDIM, DDIM);
  gemm_big(xh, wkh, kraw, MROWS, HDIM*KDIM, DDIM);
  gemm_big(xh, wvh, vraw, MROWS, HDIM*VDIM, DDIM);
  gemm_big(xh, wf1h, f1, MROWS, VDIM, DDIM);
  gemm_small(xh, wbh, betar, MROWS, HDIM, DDIM);
  gemm_big(xh, wg1h, g1, MROWS, VDIM, DDIM);
  cvt(f1, f1h, (size_t)MROWS*VDIM);
  cvt(g1, g1h, (size_t)MROWS*VDIM);
  gemm_big(f1h, wf2h, graw, MROWS, HDIM*KDIM, VDIM);
  gemm_big(g1h, wg2h, gater, MROWS, HDIM*VDIM, VDIM);

  // 3) conv + silu (+l2norm for q/k)
  const unsigned nbth = BDIM * TDIM * HDIM;
  conv_silu<<<nbth, 128, 0, stream>>>(qraw, qcw, qn, 1);
  conv_silu<<<nbth, 128, 0, stream>>>(kraw, kcw, kn, 1);
  conv_silu<<<nbth, 128, 0, stream>>>(vraw, vcw, vn, 0);

  // 4) decay precompute (in place on graw)
  decay_kernel<<<(unsigned)((NE_HK + 255) / 256), 256, 0, stream>>>(graw, A_log, dtb, (int)NE_HK);

  // 5) sequential delta-rule scan
  kda_scan<<<BDIM * HDIM, 128, 0, stream>>>(qn, kn, vn, graw, betar, ybuf);

  // 6) rmsnorm + gate epilogue -> f16
  epilogue<<<nbth, 128, 0, stream>>>(ybuf, gater, bg, onw, y2h);

  // 7) output projection (WMMA) -> d_out
  gemm_big(y2h, wouth, out, MROWS, DDIM, HDIM*VDIM);
}
